// GCN4Layers_62526133895431
// MI455X (gfx1250) — matrix operand order repaired and checked by reading, not compiled, not run on hardware
//
#include <hip/hip_runtime.h>

typedef __attribute__((ext_vector_type(16))) _Float16 v16h;
typedef __attribute__((ext_vector_type(8)))  float    v8f;

// ---------------------------------------------------------------- utilities

// n must be a multiple of 4 (all call sites guarantee this)
__global__ void fill_zero4_kernel(float4* __restrict__ p, int n4) {
    int i = blockIdx.x * blockDim.x + threadIdx.x;
    if (i < n4) p[i] = make_float4(0.0f, 0.0f, 0.0f, 0.0f);
}

__global__ void degree_kernel(const int* __restrict__ src, const int* __restrict__ dst,
                              float* __restrict__ deg_out, float* __restrict__ deg_in, int E) {
    int e = blockIdx.x * blockDim.x + threadIdx.x;
    if (e < E) {
        atomicAdd(&deg_out[src[e]], 1.0f);
        atomicAdd(&deg_in[dst[e]], 1.0f);
    }
}

__global__ void norm_kernel(const float* __restrict__ deg_out, const float* __restrict__ deg_in,
                            float* __restrict__ nrm_out, float* __restrict__ nrm_in, int n) {
    int i = blockIdx.x * blockDim.x + threadIdx.x;
    if (i < n) {
        nrm_out[i] = rsqrtf(fmaxf(deg_out[i], 1.0f));
        nrm_in[i]  = rsqrtf(fmaxf(deg_in[i],  1.0f));
    }
}

// agg[dst] += h[src] * norm_out[src], feature width F (multiple of 4).
// Consecutive lanes cover consecutive feature quads of one edge -> coalesced
// loads and cacheline-clustered global_atomic_add_f32.
__global__ void scatter_kernel(const float* __restrict__ h,
                               const int* __restrict__ src, const int* __restrict__ dst,
                               const float* __restrict__ nrm_out,
                               float* __restrict__ agg, int E, int F) {
    int vecs = F >> 2;
    int idx = blockIdx.x * blockDim.x + threadIdx.x;
    if (idx >= E * vecs) return;
    int e = idx / vecs;
    int f = (idx - e * vecs) << 2;
    int s = src[e], d = dst[e];
    float sc = nrm_out[s];
    const float4 v = *reinterpret_cast<const float4*>(h + (size_t)s * F + f);
    float* o = agg + (size_t)d * F + f;
    atomicAdd(o + 0, v.x * sc);
    atomicAdd(o + 1, v.y * sc);
    atomicAdd(o + 2, v.z * sc);
    atomicAdd(o + 3, v.w * sc);
}

// h = [relu](h * norm_in[row] + bias[col]), in place
__global__ void eltwise_kernel(float* __restrict__ h, const float* __restrict__ nrm_in,
                               const float* __restrict__ bias, int n, int F, int do_relu) {
    int idx = blockIdx.x * blockDim.x + threadIdx.x;
    if (idx >= n * F) return;
    int m = idx / F;
    int f = idx - m * F;
    float v = h[idx] * nrm_in[m] + bias[f];
    h[idx] = do_relu ? fmaxf(v, 0.0f) : v;
}

// ------------------------------------------------- WMMA GEMM (f16 in, f32 acc)
// out[M x Nreal] = [relu]( (A * rowScale?) [M x KDIM] @ W [KDIM x Nreal] + bias? )
// One wave32 per 16x16 output tile; W staged in LDS as f16, padded to NPAD cols.
template <int KDIM, int NPAD>
__global__ __launch_bounds__(128) void wmma_gemm_kernel(
        const float* __restrict__ A, const float* __restrict__ W,
        const float* __restrict__ bias, const float* __restrict__ rowScale,
        float* __restrict__ out, int M, int Nreal, int do_relu) {
    __shared__ _Float16 sW[KDIM * NPAD];
    for (int i = threadIdx.x; i < KDIM * NPAD; i += blockDim.x) {
        int k = i / NPAD;
        int n = i - k * NPAD;
        sW[i] = (n < Nreal) ? (_Float16)W[k * Nreal + n] : (_Float16)0.0f;
    }
    __syncthreads();

    const int lane   = threadIdx.x & 31;
    const int wave   = threadIdx.x >> 5;
    const int laneLo = lane & 15;
    const int laneHi = lane >> 4;           // 0 or 1
    const int ntiles = NPAD / 16;
    const int mtiles = (M + 15) >> 4;

    int tile = blockIdx.x * 4 + wave;       // wave-uniform guard: EXEC all-1s inside
    if (tile >= mtiles * ntiles) return;
    int mtile = tile / ntiles;
    int ntile = tile - mtile * ntiles;

    int row  = mtile * 16 + laneLo;
    if (row >= M) row = M - 1;              // padded rows: load safe, store guarded
    const int ncol = ntile * 16 + laneLo;
    const float rs = rowScale ? rowScale[row] : 1.0f;

    v8f c = {};
    for (int kb = 0; kb < KDIM; kb += 32) {
        // A fragment: lane holds row, K = kb + laneHi*16 + {0..15}
        const float* ap = A + (size_t)row * KDIM + kb + laneHi * 16;
        float4 a0 = *reinterpret_cast<const float4*>(ap + 0);
        float4 a1 = *reinterpret_cast<const float4*>(ap + 4);
        float4 a2 = *reinterpret_cast<const float4*>(ap + 8);
        float4 a3 = *reinterpret_cast<const float4*>(ap + 12);
        float af[16] = {a0.x, a0.y, a0.z, a0.w, a1.x, a1.y, a1.z, a1.w,
                        a2.x, a2.y, a2.z, a2.w, a3.x, a3.y, a3.z, a3.w};
        v16h a;
#pragma unroll
        for (int j = 0; j < 16; ++j) a[j] = (_Float16)(af[j] * rs);

        // B fragment from LDS: lane holds column ncol, K = kb + laneHi*16 + j
        v16h b;
        const int kb2 = kb + laneHi * 16;
#pragma unroll
        for (int j = 0; j < 16; ++j) b[j] = sW[(kb2 + j) * NPAD + ncol];

        c = __builtin_amdgcn_wmma_f32_16x16x32_f16(
                /*neg_a=*/false, a, /*neg_b=*/false, b,
                /*c_mod=*/(short)0, c, /*reuse_a=*/false, /*reuse_b=*/false);
    }

    // Epilogue: hoist column guard + bias load; additive row addressing.
    // C/D layout: VGPR r <-> row mtile*16 + r + 8*laneHi, col = ncol.
    if (ncol < Nreal) {
        const float bval = bias ? bias[ncol] : 0.0f;
        const int m0 = mtile * 16 + laneHi * 8;
        float* op = out + (size_t)m0 * Nreal + ncol;
        if (m0 + 7 < M) {
#pragma unroll
            for (int r = 0; r < 8; ++r) {
                float v = c[r] + bval;
                if (do_relu) v = fmaxf(v, 0.0f);
                op[(size_t)r * Nreal] = v;
            }
        } else {
#pragma unroll
            for (int r = 0; r < 8; ++r) {
                if (m0 + r < M) {
                    float v = c[r] + bval;
                    if (do_relu) v = fmaxf(v, 0.0f);
                    op[(size_t)r * Nreal] = v;
                }
            }
        }
    }
}

// ---------------------------------------------------------------- pooling

__global__ void pool_sum_kernel(const float* __restrict__ h4, const int* __restrict__ gid,
                                float* __restrict__ sums, float* __restrict__ counts, int n) {
    int i = blockIdx.x * blockDim.x + threadIdx.x;
    if (i >= n) return;
    int g = gid[i];
    float4 v = *reinterpret_cast<const float4*>(h4 + (size_t)i * 4);
    atomicAdd(&sums[g * 4 + 0], v.x);
    atomicAdd(&sums[g * 4 + 1], v.y);
    atomicAdd(&sums[g * 4 + 2], v.z);
    atomicAdd(&sums[g * 4 + 3], v.w);
    atomicAdd(&counts[g], 1.0f);
}

__global__ void pool_div_kernel(const float* __restrict__ sums, const float* __restrict__ counts,
                                float* __restrict__ out, int G) {
    int i = blockIdx.x * blockDim.x + threadIdx.x;
    if (i < G * 4) out[i] = sums[i] / fmaxf(counts[i >> 2], 1.0f);
}

// ---------------------------------------------------------------- launch

static inline int cdiv(long long a, int b) { return (int)((a + b - 1) / b); }

static inline void fill0(float* p, long long n, hipStream_t stream) {
    // n is always a multiple of 4 here
    fill_zero4_kernel<<<cdiv(n / 4, 256), 256, 0, stream>>>((float4*)p, (int)(n / 4));
}

extern "C" void kernel_launch(void* const* d_in, const int* in_sizes, int n_in,
                              void* d_out, int out_size, void* d_ws, size_t ws_size,
                              hipStream_t stream) {
    const float* x  = (const float*)d_in[0];
    const float* W1 = (const float*)d_in[1];
    const float* b1 = (const float*)d_in[2];
    const float* W2 = (const float*)d_in[3];
    const float* b2 = (const float*)d_in[4];
    const float* W3 = (const float*)d_in[5];
    const float* b3 = (const float*)d_in[6];
    const float* W4 = (const float*)d_in[7];
    const float* b4 = (const float*)d_in[8];
    const int* src  = (const int*)d_in[9];
    const int* dst  = (const int*)d_in[10];
    const int* gid  = (const int*)d_in[11];
    float* out = (float*)d_out;

    const int N = in_sizes[0] / 64;   // 100000 nodes
    const int E = in_sizes[9];        // 800000 edges
    const int G = out_size / 4;       // 500 graphs

    // workspace layout (floats)
    float* ws      = (float*)d_ws;
    float* deg_out = ws;
    float* deg_in  = deg_out + (size_t)N;
    float* nrm_out = deg_in  + (size_t)N;
    float* nrm_in  = nrm_out + (size_t)N;
    float* bufA    = nrm_in  + (size_t)N;          // up to N x 64
    float* bufB    = bufA    + (size_t)N * 64;     // up to N x 128
    float* sums    = bufB    + (size_t)N * 128;    // G x 4
    // counts = sums + G*4 (zeroed together with sums; G*5 is a multiple of 4)
    float* counts  = sums    + (size_t)G * 4;
    size_t need = ((size_t)N * (4 + 64 + 128) + (size_t)G * 5) * sizeof(float);
    if (ws_size < need) return;

    const int BT = 256;
    const int mtiles = (N + 15) / 16;

    // degrees -> norms
    fill0(deg_out, 2LL * N, stream);
    degree_kernel<<<cdiv(E, BT), BT, 0, stream>>>(src, dst, deg_out, deg_in, E);
    norm_kernel<<<cdiv(N, BT), BT, 0, stream>>>(deg_out, deg_in, nrm_out, nrm_in, N);

    // layer 1 (64 -> 128): aggregate first, then GEMM(norm_in, +b1, relu)
    fill0(bufA, (long long)N * 64, stream);
    scatter_kernel<<<cdiv((long long)E * 16, BT), BT, 0, stream>>>(x, src, dst, nrm_out, bufA, E, 64);
    wmma_gemm_kernel<64, 128><<<cdiv((long long)mtiles * 8, 4), 128, 0, stream>>>(
        bufA, W1, b1, nrm_in, bufB, N, 128, 1);

    // layer 2 (128 -> 64): GEMM first, then aggregate, then eltwise(norm_in,+b2,relu)
    wmma_gemm_kernel<128, 64><<<cdiv((long long)mtiles * 4, 4), 128, 0, stream>>>(
        bufB, W2, nullptr, nullptr, bufA, N, 64, 0);
    fill0(bufB, (long long)N * 64, stream);
    scatter_kernel<<<cdiv((long long)E * 16, BT), BT, 0, stream>>>(bufA, src, dst, nrm_out, bufB, E, 64);
    eltwise_kernel<<<cdiv((long long)N * 64, BT), BT, 0, stream>>>(bufB, nrm_in, b2, N, 64, 1);

    // layer 3 (64 -> 32)
    wmma_gemm_kernel<64, 32><<<cdiv((long long)mtiles * 2, 4), 128, 0, stream>>>(
        bufB, W3, nullptr, nullptr, bufA, N, 32, 0);
    fill0(bufB, (long long)N * 32, stream);
    scatter_kernel<<<cdiv((long long)E * 8, BT), BT, 0, stream>>>(bufA, src, dst, nrm_out, bufB, E, 32);
    eltwise_kernel<<<cdiv((long long)N * 32, BT), BT, 0, stream>>>(bufB, nrm_in, b3, N, 32, 1);

    // layer 4 (32 -> 4), padded to one 16-col tile, no relu
    wmma_gemm_kernel<32, 16><<<cdiv((long long)mtiles, 4), 128, 0, stream>>>(
        bufB, W4, nullptr, nullptr, bufA, N, 4, 0);
    fill0(bufB, (long long)N * 4, stream);
    scatter_kernel<<<cdiv((long long)E, BT), BT, 0, stream>>>(bufA, src, dst, nrm_out, bufB, E, 4);
    eltwise_kernel<<<cdiv((long long)N * 4, BT), BT, 0, stream>>>(bufB, nrm_in, b4, N, 4, 0);

    // graph mean pooling
    fill0(sums, (long long)G * 5, stream);
    pool_sum_kernel<<<cdiv(N, BT), BT, 0, stream>>>(bufB, gid, sums, counts, N);
    pool_div_kernel<<<cdiv((long long)G * 4, BT), BT, 0, stream>>>(sums, counts, out, G);
}
